// BackboneDEARI_84808424227323
// MI455X (gfx1250) — matrix operand order repaired and checked by reading, not compiled, not run on hardware
//
#include <hip/hip_runtime.h>

// ---------------------------------------------------------------------------
// DEARI / BRITS recurrent imputation for MI455X (gfx1250), wave32 + WMMA bf16.
// B=512, T=64, F=256, H=1024.
//
// Design (compute-bound: ~356 GFLOP vs ~9us of HBM traffic at 23.3 TB/s):
//  * One persistent kernel; 32 workgroups x 512 threads (16 waves).
//  * Each WG owns a 16-row batch slab (WMMA M=16); h state lives in LDS
//    across all 64 timesteps -> no grid-wide sync ever needed.
//  * All 6 per-step GEMMs run on v_wmma_f32_16x16x32_bf16; weights are
//    converted once to bf16 in workspace (10.9 MB, resident in 192MB L2).
//  * GRU computed in 4 column-groups per wave, r->z->n fused per group so
//    only ~16 persistent VGPRs are live (previous version spilled 96+).
//  * LDS kept at ~144.1 KB (CU-mode-safe); loss partials deterministic.
// ---------------------------------------------------------------------------

#define BB 512
#define TT 64
#define FF 256
#define HH 1024

typedef unsigned short ushort_t;
typedef unsigned int   uint_t;
typedef __attribute__((ext_vector_type(16))) __bf16 v16bf;
typedef __attribute__((ext_vector_type(8)))  float  v8f;
typedef __attribute__((ext_vector_type(8)))  unsigned int v8u;

// ws layout (ushort units for bf16 weights)
#define OFF_WDH 0u          // 1024x256
#define OFF_WH  262144u     // 256x1024
#define OFF_WFR 524288u     // 256x256 (diag zeroed)
#define OFF_WWC 589824u     // 256x512
#define OFF_WIH 720896u     // 3072x512
#define OFF_WHH 2293760u    // 3072x1024
#define LOSS_BYTE_OFF 10878976u

// d_out layout (f32 elements)
#define OUT_IMP  0u
#define OUT_REC  8388608u      // B*T*F
#define OUT_HFIN 16777216u     // 2*B*T*F
#define OUT_XLOSS 17301504u    // + B*H
#define OUT_KL    17301505u

__device__ __forceinline__ ushort_t f2bf(float f) {
  uint_t u = __builtin_bit_cast(uint_t, f);
  u += 0x7FFFu + ((u >> 16) & 1u);
  return (ushort_t)(u >> 16);
}

// A-matrix fragment: 16-bit A 16x32 layout (ISA 7.12.2): lane holds row
// (lane&15); K chunks at kbase=(lane>>4)*8 : [kbase..kbase+7], [kbase+16..+23].
__device__ __forceinline__ v16bf load_afrag(const ushort_t* p) {
  uint4 lo = *(const uint4*)(p);
  uint4 hi = *(const uint4*)(p + 16);
  v8u t = {lo.x, lo.y, lo.z, lo.w, hi.x, hi.y, hi.z, hi.w};
  return __builtin_bit_cast(v16bf, t);
}
// B-matrix fragment: 16-bit B 32x16: lane holds col (lane&15) = weight row,
// K chunks at kbase=(lane>>4)*8 : [kbase..kbase+7], [kbase+16..+23].
__device__ __forceinline__ v16bf load_bfrag(const ushort_t* p) {
  uint4 lo = *(const uint4*)(p);
  uint4 hi = *(const uint4*)(p + 16);
  v8u t = {lo.x, lo.y, lo.z, lo.w, hi.x, hi.y, hi.z, hi.w};
  return __builtin_bit_cast(v16bf, t);
}

// One 16x16 output tile of A[16,kTot] @ W[nt*16..+15, kTot]^T, K-major bf16.
__device__ __forceinline__ v8f gemm16(const ushort_t* Abase, int ldA,
                                      const ushort_t* Wbase, int ldW,
                                      int nt, int kTot, int lane) {
  const int half = lane >> 4;
  const ushort_t* ap = Abase + (lane & 15) * ldA + half * 8;
  const ushort_t* bp = Wbase + (size_t)(nt * 16 + (lane & 15)) * ldW + half * 8;
  v8f acc = {0.f, 0.f, 0.f, 0.f, 0.f, 0.f, 0.f, 0.f};
#pragma unroll 4
  for (int k = 0; k < kTot; k += 32) {
    v16bf a = load_afrag(ap + k);
    v16bf b = load_bfrag(bp + k);
    acc = __builtin_amdgcn_wmma_f32_16x16x32_bf16(false, a, false, b,
                                                  (short)0, acc, false, false);
  }
  return acc;
}

__device__ __forceinline__ float sigmoidf(float g) {
  return 1.f / (1.f + __expf(-g));
}

// ---------------------------------------------------------------------------
__global__ void conv_bf16_k(const float* __restrict__ src,
                            ushort_t* __restrict__ dst, int n) {
  int i = blockIdx.x * blockDim.x + threadIdx.x;
  if (i < n) dst[i] = f2bf(src[i]);
}
__global__ void conv_bf16_offdiag_k(const float* __restrict__ src,
                                    ushort_t* __restrict__ dst, int n, int dim) {
  int i = blockIdx.x * blockDim.x + threadIdx.x;
  if (i < n) {
    int r = i / dim, c = i - r * dim;
    dst[i] = f2bf((r == c) ? 0.f : src[i]);
  }
}

// ---------------------------------------------------------------------------
__global__ void __launch_bounds__(512, 1)
deari_main(const float* __restrict__ x, const float* __restrict__ mask,
           const float* __restrict__ deltas, const float* __restrict__ h0,
           const float* __restrict__ bdh, const float* __restrict__ Wdx,
           const float* __restrict__ bdx, const float* __restrict__ bh,
           const float* __restrict__ bfr, const float* __restrict__ bwc,
           const float* __restrict__ bih, const float* __restrict__ bhh,
           const ushort_t* __restrict__ wbf, float* __restrict__ out,
           float* __restrict__ wsloss) {
  __shared__ float    sH[16 * HH];      // 64 KB  f32 hidden state
  __shared__ ushort_t sHbf[16 * HH];    // 32 KB  bf16 mirror of decayed h
  __shared__ ushort_t sA[16 * 1024];    // 32 KB  staging for A matrices
  __shared__ float    sXh[16 * FF];     // 16 KB  x_h
  __shared__ float    sRed[32];

  const ushort_t* Wdh_bf = wbf + OFF_WDH;
  const ushort_t* Wh_bf  = wbf + OFF_WH;
  const ushort_t* Wfr_bf = wbf + OFF_WFR;
  const ushort_t* Wwc_bf = wbf + OFF_WWC;
  const ushort_t* Wih_bf = wbf + OFF_WIH;
  const ushort_t* Whh_bf = wbf + OFF_WHH;

  const int tid = threadIdx.x, wv = tid >> 5, lane = tid & 31;
  const int wg = blockIdx.x, mb = wg * 16;
  const int ln = lane & 15, mb8 = (lane >> 4) * 8;

  for (int i = tid; i < 16 * HH; i += 512) {
    int m = i >> 10, j = i & (HH - 1);
    sH[i] = h0[(size_t)(mb + m) * HH + j];
  }
  __syncthreads();

  float* out_imp = out + OUT_IMP;
  float* out_rec = out + OUT_REC;

  for (int t = 0; t < TT; ++t) {
    // -------- stage A: deltas -> sA[:, 0:256] bf16 (+ prefetch next step)
    for (int i = tid; i < 16 * FF; i += 512) {
      int m = i >> 8, f = i & (FF - 1);
      size_t g = ((size_t)(mb + m) * TT + t) * FF + f;
      sA[m * 1024 + f] = f2bf(deltas[g]);
      if (t + 1 < TT) {
        __builtin_prefetch(&deltas[g + FF], 0, 0);
        __builtin_prefetch(&x[g + FF], 0, 0);
        __builtin_prefetch(&mask[g + FF], 0, 0);
      }
    }
    __syncthreads();

    // -------- GEMM1: gamma_h = exp(-relu(d@Wdh^T+bdh)); h *= gamma_h
#pragma unroll 1
    for (int i = 0; i < 4; ++i) {
      int nt = wv + 16 * i;
      v8f acc = gemm16(sA, 1024, Wdh_bf, FF, nt, FF, lane);
      int col = nt * 16 + ln;
      float bb = bdh[col];
#pragma unroll
      for (int r = 0; r < 8; ++r) {
        int m = mb8 + r;
        float g = __expf(-fmaxf(acc[r] + bb, 0.f));
        float hd = sH[m * HH + col] * g;
        sH[m * HH + col] = hd;
        sHbf[m * HH + col] = f2bf(hd);
      }
    }
    __syncthreads();

    // -------- GEMM2: x_h = h@Wh^T + bh
    {
      v8f acc = gemm16(sHbf, HH, Wh_bf, HH, wv, HH, lane);
      int col = wv * 16 + ln;
      float bb = bh[col];
#pragma unroll
      for (int r = 0; r < 8; ++r) sXh[(mb8 + r) * FF + col] = acc[r] + bb;
    }
    __syncthreads();

    // -------- elementwise: x_r, gamma_x, pack [x_r | gamma_x | m] into sA
    for (int i = tid; i < 16 * FF; i += 512) {
      int m = i >> 8, f = i & (FF - 1);
      size_t g = ((size_t)(mb + m) * TT + t) * FF + f;
      float xv = x[g], mv = mask[g], dv = deltas[g];
      float xr = mv * xv + (1.f - mv) * sXh[m * FF + f];
      float gx = __expf(-fmaxf(dv * Wdx[f * FF + f] + bdx[f], 0.f));
      sA[m * 1024 + f] = f2bf(xr);
      sA[m * 1024 + 256 + f] = f2bf(gx);
      sA[m * 1024 + 512 + f] = f2bf(mv);
    }
    __syncthreads();

    // -------- GEMM3: xu = x_r@Wfr_off^T ; GEMM4: beta = [gx,m]@Wwc^T
    v8f axu = gemm16(sA, 1024, Wfr_bf, FF, wv, FF, lane);
    v8f abt = gemm16(sA + 256, 1024, Wwc_bf, 2 * FF, wv, 2 * FF, lane);
    __syncthreads();  // everyone done reading sA before epilogue rewrites it

    // -------- combine epilogue + loss + outputs + rnn_in = [x_imp | m]
    float lnum = 0.f, lden = 0.f;
    {
      int col = wv * 16 + ln;
      float bxu = bfr[col], bbt = bwc[col];
#pragma unroll
      for (int r = 0; r < 8; ++r) {
        int m = mb8 + r;
        float xu = axu[r] + bxu;
        float beta = abt[r] + bbt;
        float xh = sXh[m * FF + col];
        size_t g = ((size_t)(mb + m) * TT + t) * FF + col;
        float xv = x[g], mv = mask[g];
        float xc = beta * xu + (1.f - beta) * xh;
        lnum += fabsf(xc - xv) * mv;
        lden += mv;
        float xi = mv * xv + (1.f - mv) * xc;
        out_imp[g] = xi;
        out_rec[g] = xc;
        sA[m * 1024 + col] = f2bf(xi);
        sA[m * 1024 + 256 + col] = f2bf(mv);
      }
    }
#pragma unroll
    for (int off = 16; off; off >>= 1) {
      lnum += __shfl_xor(lnum, off);
      lden += __shfl_xor(lden, off);
    }
    if (lane == 0) { sRed[2 * wv] = lnum; sRed[2 * wv + 1] = lden; }
    __syncthreads();
    if (tid == 0) {
      float sn = 0.f, sd = 0.f;
      for (int w = 0; w < 16; ++w) { sn += sRed[2 * w]; sd += sRed[2 * w + 1]; }
      wsloss[t * 32 + wg] = sn;
      wsloss[2048 + t * 32 + wg] = sd;
    }

    // -------- GRU: 4 column-groups per wave; r -> z -> n fused per group.
    // Wave w's group i covers gate columns j = (w+16i)*16 + ln (+8 rows/lane).
    // All gate data is lane-private: no barriers, no LDS, ~16 persistent VGPRs.
#pragma unroll 1
    for (int i = 0; i < 4; ++i) {
      const int ntr = wv + 16 * i;        // r tile   (cols    0..1023 of 3H)
      const int ntz = 64 + ntr;           // z tile   (cols 1024..2047)
      const int ntn = 128 + ntr;          // n tile   (cols 2048..3071)
      const int j = ntr * 16 + ln;        // column within gate (0..1023)

      v8f rg, zg;
      {
        v8f a1 = gemm16(sA, 1024, Wih_bf, 2 * FF, ntr, 2 * FF, lane);
        v8f a2 = gemm16(sHbf, HH, Whh_bf, HH, ntr, HH, lane);
        float bi = bih[ntr * 16 + ln], bhv = bhh[ntr * 16 + ln];
#pragma unroll
        for (int r = 0; r < 8; ++r) rg[r] = sigmoidf(a1[r] + bi + a2[r] + bhv);
      }
      {
        v8f a1 = gemm16(sA, 1024, Wih_bf, 2 * FF, ntz, 2 * FF, lane);
        v8f a2 = gemm16(sHbf, HH, Whh_bf, HH, ntz, HH, lane);
        float bi = bih[ntz * 16 + ln], bhv = bhh[ntz * 16 + ln];
#pragma unroll
        for (int r = 0; r < 8; ++r) zg[r] = sigmoidf(a1[r] + bi + a2[r] + bhv);
      }
      {
        v8f a1 = gemm16(sA, 1024, Wih_bf, 2 * FF, ntn, 2 * FF, lane);
        v8f a2 = gemm16(sHbf, HH, Whh_bf, HH, ntn, HH, lane);
        float bi = bih[ntn * 16 + ln], bhv = bhh[ntn * 16 + ln];
#pragma unroll
        for (int r = 0; r < 8; ++r) {
          int m = mb8 + r;
          float nn = tanhf(a1[r] + bi + rg[r] * (a2[r] + bhv));
          float hold = sH[m * HH + j];
          sH[m * HH + j] = (1.f - zg[r]) * nn + zg[r] * hold;
        }
      }
    }
    __syncthreads();  // h writes visible before next step's GEMM1
  }

  // -------- final hidden state
  float* out_h = out + OUT_HFIN;
  for (int i = tid; i < 16 * HH; i += 512) {
    int m = i >> 10, j = i & (HH - 1);
    out_h[(size_t)(mb + m) * HH + j] = sH[i];
  }
}

// ---------------------------------------------------------------------------
__global__ void finalize_k(const float* __restrict__ wsloss,
                           float* __restrict__ out) {
  __shared__ float s[64];
  int t = threadIdx.x;  // 64 threads
  float sn = 0.f, sd = 0.f;
  for (int w = 0; w < 32; ++w) {
    sn += wsloss[t * 32 + w];
    sd += wsloss[2048 + t * 32 + w];
  }
  s[t] = sn / (sd + 1e-12f);
  __syncthreads();
  if (t == 0) {
    float acc = 0.f;
    for (int i = 0; i < 64; ++i) acc += s[i];
    out[OUT_XLOSS] = acc;
    out[OUT_KL] = 0.f;
  }
}

// ---------------------------------------------------------------------------
extern "C" void kernel_launch(void* const* d_in, const int* in_sizes, int n_in,
                              void* d_out, int out_size, void* d_ws,
                              size_t ws_size, hipStream_t stream) {
  (void)in_sizes; (void)n_in; (void)out_size; (void)ws_size;
  const float* x      = (const float*)d_in[0];
  const float* mask   = (const float*)d_in[1];
  const float* deltas = (const float*)d_in[2];
  const float* h0     = (const float*)d_in[3];
  const float* Wdh    = (const float*)d_in[4];
  const float* bdh    = (const float*)d_in[5];
  const float* Wdx    = (const float*)d_in[6];
  const float* bdx    = (const float*)d_in[7];
  const float* Wh     = (const float*)d_in[8];
  const float* bh     = (const float*)d_in[9];
  const float* Wfr    = (const float*)d_in[10];
  const float* bfr    = (const float*)d_in[11];
  const float* Wwc    = (const float*)d_in[12];
  const float* bwc    = (const float*)d_in[13];
  const float* Wih    = (const float*)d_in[14];
  const float* bih    = (const float*)d_in[15];
  const float* Whh    = (const float*)d_in[16];
  const float* bhh    = (const float*)d_in[17];

  ushort_t* wbf = (ushort_t*)d_ws;
  float* wsloss = (float*)((char*)d_ws + LOSS_BYTE_OFF);
  float* out = (float*)d_out;

  auto cdiv = [](int a, int b) { return (a + b - 1) / b; };
  conv_bf16_k<<<cdiv(262144, 256), 256, 0, stream>>>(Wdh, wbf + OFF_WDH, 262144);
  conv_bf16_k<<<cdiv(262144, 256), 256, 0, stream>>>(Wh, wbf + OFF_WH, 262144);
  conv_bf16_offdiag_k<<<cdiv(65536, 256), 256, 0, stream>>>(Wfr, wbf + OFF_WFR,
                                                            65536, 256);
  conv_bf16_k<<<cdiv(131072, 256), 256, 0, stream>>>(Wwc, wbf + OFF_WWC, 131072);
  conv_bf16_k<<<cdiv(1572864, 256), 256, 0, stream>>>(Wih, wbf + OFF_WIH, 1572864);
  conv_bf16_k<<<cdiv(3145728, 256), 256, 0, stream>>>(Whh, wbf + OFF_WHH, 3145728);

  deari_main<<<32, 512, 0, stream>>>(x, mask, deltas, h0, bdh, Wdx, bdx, bh,
                                     bfr, bwc, bih, bhh, wbf, out, wsloss);
  finalize_k<<<1, 64, 0, stream>>>(wsloss, out);
}
